// LocalTransformerBlock_7816840479391
// MI455X (gfx1250) — matrix operand order repaired and checked by reading, not compiled, not run on hardware
//
#include <hip/hip_runtime.h>

typedef unsigned short u16;
typedef __bf16 bf16x16 __attribute__((ext_vector_type(16)));
typedef float f32x8 __attribute__((ext_vector_type(8)));
typedef int i32x4 __attribute__((ext_vector_type(4)));

union Frag {
    bf16x16 v;
    uint4 q[2];
};

__device__ inline u16 f2b(float f) {
    union { float f; unsigned int u; } x;
    x.f = f;
    unsigned int r = x.u + 0x7fffu + ((x.u >> 16) & 1u);
    return (u16)(r >> 16);
}
__device__ inline float b2f(u16 h) {
    union { unsigned int u; float f; } x;
    x.u = ((unsigned int)h) << 16;
    return x.f;
}

__device__ inline f32x8 wmma_bf16(bf16x16 a, bf16x16 b, f32x8 c) {
    return __builtin_amdgcn_wmma_f32_16x16x32_bf16(false, a, false, b, (short)0, c, false, false);
}

// ---------------------------------------------------------------------------
// Async global->LDS copy (CDNA5 GLOBAL_LOAD_ASYNC_TO_LDS_B128, ASYNCcnt).
// Builtin signature (from hipcc diagnostic): param0 = int4* in AS1 (global),
// param1 = int4* in AS3 (LDS), then imm offset and imm cpol.
// Falls back to a register-staged copy if the builtin is unavailable; the
// barrier schedule below is correct for both paths.
// ---------------------------------------------------------------------------
#if defined(__gfx1250__) && __has_builtin(__builtin_amdgcn_global_load_async_to_lds_b128)
#define ATT_HAS_ASYNC_LDS 1
#else
#define ATT_HAS_ASYNC_LDS 0
#endif

#if ATT_HAS_ASYNC_LDS
typedef __attribute__((address_space(1))) i32x4 as1_i32x4;
typedef __attribute__((address_space(3))) i32x4 as3_i32x4;
#endif

__device__ inline void cp_b128(const u16* g, u16* l) {
#if ATT_HAS_ASYNC_LDS
    __builtin_amdgcn_global_load_async_to_lds_b128(
        (as1_i32x4*)g, (as3_i32x4*)l, 0, 0);
#else
    *(uint4*)l = *(const uint4*)g;
#endif
}

__device__ inline void async_wait_all() {
#if ATT_HAS_ASYNC_LDS
#if __has_builtin(__builtin_amdgcn_s_wait_asynccnt)
    __builtin_amdgcn_s_wait_asynccnt(0);
#else
    asm volatile("s_wait_asynccnt 0x0" ::: "memory");
#endif
#endif
}

// ---------------------------------------------------------------------------
// Weight convert: src f32 [K, N] row-major (row stride srcLd) ->
// dst bf16 [Nrows, Kpad] (transposed, zero padded).
// ---------------------------------------------------------------------------
__global__ __launch_bounds__(256) void convT_kernel(const float* __restrict__ src, int srcLd,
                                                    int K, int N,
                                                    u16* __restrict__ dst, int Kpad, int Nrows) {
    size_t idx = (size_t)blockIdx.x * 256 + threadIdx.x;
    int n = (int)(idx / Kpad);
    int kk = (int)(idx % Kpad);
    if (n >= Nrows) return;
    float v = (kk < K && n < N) ? src[(size_t)kk * srcLd + n] : 0.0f;
    dst[(size_t)n * Kpad + kk] = f2b(v);
}

// ---------------------------------------------------------------------------
// LayerNorm over D=1024: one block per row, 256 threads x 4 elements.
// ---------------------------------------------------------------------------
__global__ __launch_bounds__(256) void ln_kernel(const float* __restrict__ x,
                                                 const float* __restrict__ gamma,
                                                 const float* __restrict__ beta, int has_beta,
                                                 u16* __restrict__ out) {
    int tid = threadIdx.x;
    int lane = tid & 31;
    int wid = tid >> 5;
    size_t row = blockIdx.x;
    const float4* xr = reinterpret_cast<const float4*>(x + row * 1024);
    float4 v = xr[tid];
    float s1 = v.x + v.y + v.z + v.w;
    float s2 = v.x * v.x + v.y * v.y + v.z * v.z + v.w * v.w;
    for (int off = 16; off > 0; off >>= 1) {
        s1 += __shfl_xor(s1, off, 32);
        s2 += __shfl_xor(s2, off, 32);
    }
    __shared__ float sh1[8], sh2[8];
    if (lane == 0) { sh1[wid] = s1; sh2[wid] = s2; }
    __syncthreads();
    float S1 = 0.f, S2 = 0.f;
    for (int i = 0; i < 8; ++i) { S1 += sh1[i]; S2 += sh2[i]; }
    float mean = S1 * (1.0f / 1024.0f);
    float var = S2 * (1.0f / 1024.0f) - mean * mean;
    float rs = rsqrtf(var + 1e-5f);
    const float4 g = reinterpret_cast<const float4*>(gamma)[tid];
    float4 b = make_float4(0.f, 0.f, 0.f, 0.f);
    if (has_beta) b = reinterpret_cast<const float4*>(beta)[tid];
    float ys[4];
    ys[0] = (v.x - mean) * rs * g.x + b.x;
    ys[1] = (v.y - mean) * rs * g.y + b.y;
    ys[2] = (v.z - mean) * rs * g.z + b.z;
    ys[3] = (v.w - mean) * rs * g.w + b.w;
    u16* o = out + row * 1024 + tid * 4;
    for (int j = 0; j < 4; ++j) o[j] = f2b(ys[j]);
}

// ---------------------------------------------------------------------------
// Generic bf16 WMMA GEMM. C[M,Ntot] = A[M,K](bf16) x BT[Ntot,K](bf16,transposed)
// EPI: 0 = store f32, 1 = store bf16, 2 = store f32 with residual add from R.
// Block tile 128x128x32; 8 waves -> each wave 64x32 (4x2 wmma tiles).
// Double-buffered: async global->LDS copies of tile k+1 overlap WMMAs of
// tile k; one barrier + one s_wait_asynccnt per K-step.
// ---------------------------------------------------------------------------
template <int EPI>
__global__ __launch_bounds__(256) void gemm_bf16_kernel(
    const u16* __restrict__ A, int lda,
    const u16* __restrict__ BT, int ldb,
    float* __restrict__ Cf, u16* __restrict__ Cb, int ldc,
    const float* __restrict__ R,
    int M, int Ntot, int K) {
    __shared__ __align__(16) u16 sA[2][128 * 40];
    __shared__ __align__(16) u16 sB[2][128 * 40];

    int tid = threadIdx.x;
    int lane = tid & 31;
    int wave = tid >> 5;
    int l15 = lane & 15;
    int wm = (wave >> 2) * 64;   // wave M offset within block (2 rows of waves)
    int wn = (wave & 3) * 32;    // wave N offset within block (4 cols of waves)
    int m0 = blockIdx.y * 128;
    int n0 = blockIdx.x * 128;

    f32x8 acc[4][2] = {};

    int ka = (lane < 16) ? 0 : 8;   // A-fragment K base per lane-half
    int kb = (lane < 16) ? 0 : 8;   // B-fragment K base per lane-half

    auto issue_tile = [&](int k0, int buf) {
        for (int c = tid; c < 512; c += 256) {
            int r = c >> 2;
            int kk = (c & 3) << 3;
            cp_b128(A + (size_t)(m0 + r) * lda + (k0 + kk), &sA[buf][r * 40 + kk]);
            cp_b128(BT + (size_t)(n0 + r) * ldb + (k0 + kk), &sB[buf][r * 40 + kk]);
        }
    };

    int nk = K >> 5;
    issue_tile(0, 0);

    for (int kt = 0; kt < nk; ++kt) {
        int buf = kt & 1;
        async_wait_all();     // tile kt's async writes to LDS done (this wave)
        __syncthreads();      // ... and every other wave's; prev reads of buf^1 done
        if (kt + 1 < nk) issue_tile((kt + 1) << 5, buf ^ 1);

        Frag af[4], bfr[2];
        for (int mi = 0; mi < 4; ++mi) {
            const u16* p = sA[buf] + (wm + mi * 16 + l15) * 40 + ka;
            af[mi].q[0] = *(const uint4*)p;
            af[mi].q[1] = *(const uint4*)(p + 16);
        }
        for (int ni = 0; ni < 2; ++ni) {
            const u16* p = sB[buf] + (wn + ni * 16 + l15) * 40 + kb;
            bfr[ni].q[0] = *(const uint4*)p;
            bfr[ni].q[1] = *(const uint4*)(p + 16);
        }
        for (int mi = 0; mi < 4; ++mi)
            for (int ni = 0; ni < 2; ++ni)
                acc[mi][ni] = wmma_bf16(af[mi].v, bfr[ni].v, acc[mi][ni]);
    }

    int rh = (lane < 16) ? 0 : 8;
    for (int mi = 0; mi < 4; ++mi) {
        for (int ni = 0; ni < 2; ++ni) {
            int col = n0 + wn + ni * 16 + l15;
            if (col >= Ntot) continue;
            size_t rbase = (size_t)(m0 + wm + mi * 16 + rh);
            for (int i = 0; i < 8; ++i) {
                size_t idx = (rbase + i) * (size_t)ldc + col;
                float v = acc[mi][ni][i];
                if (EPI == 0) {
                    Cf[idx] = v;
                } else if (EPI == 1) {
                    Cb[idx] = f2b(v);
                } else {
                    Cf[idx] = v + R[idx];
                }
            }
        }
    }
}

// ---------------------------------------------------------------------------
// Rotary embed + head split. qkv bf16 [B*N, 3072] ->
//   q,k bf16 [B*H, N, 64]  (q scaled by DH^-0.5), v transposed bf16 [B*H, 64, N]
// ---------------------------------------------------------------------------
__global__ __launch_bounds__(256) void rotary_kernel(const u16* __restrict__ qkv,
                                                     u16* __restrict__ qg,
                                                     u16* __restrict__ kg,
                                                     u16* __restrict__ vT) {
    unsigned int idx = blockIdx.x * 256 + threadIdx.x;  // < 2*4096*16*64
    int d = idx & 63;
    int h = (idx >> 6) & 15;
    int n = (idx >> 10) & 4095;
    int b = idx >> 22;
    size_t row = (size_t)b * 4096 + n;
    size_t base = row * 3072 + h * 64;
    float qv = b2f(qkv[base + d]);
    float kv = b2f(qkv[base + 1024 + d]);
    float vv = b2f(qkv[base + 2048 + d]);
    float qp = b2f(qkv[base + (d ^ 32)]);
    float kp = b2f(qkv[base + 1024 + (d ^ 32)]);
    float rq = (d < 32) ? -qp : qp;
    float rk = (d < 32) ? -kp : kp;
    int i = d & 31;
    float inv = powf(10000.0f, -(float)i / 32.0f);
    float f = (float)n * inv;
    float sn, cs;
    sincosf(f, &sn, &cs);
    float qo = (qv * cs + rq * sn) * 0.125f;  // * DH^-0.5
    float ko = kv * cs + rk * sn;
    size_t bh = (size_t)b * 16 + h;
    qg[(bh * 4096 + n) * 64 + d] = f2b(qo);
    kg[(bh * 4096 + n) * 64 + d] = f2b(ko);
    vT[(bh * 64 + d) * 4096 + n] = f2b(vv);
}

// ---------------------------------------------------------------------------
// Windowed causal attention (look_backward=1), flash-style online softmax.
// grid = (2, nw=8, B*H=32), 256 threads (8 waves). Each wave: 32 q-rows,
// loops over the 1024-key window in 32-key chunks with WMMA for both GEMMs.
// Interior chunks take a wave-uniform fast path that skips all masking.
// Writes o bf16 [B*N, 1024] at column h*64+d.
// ---------------------------------------------------------------------------
__global__ __launch_bounds__(256) void attn_kernel(const u16* __restrict__ qg,
                                                   const u16* __restrict__ kg,
                                                   const u16* __restrict__ vT,
                                                   u16* __restrict__ og) {
    __shared__ __align__(16) u16 pls[8 * 32 * 40];

    int tid = threadIdx.x;
    int lane = tid & 31;
    int wave = tid >> 5;
    int l15 = lane & 15;
    int lh = (lane < 16) ? 0 : 1;

    int half = blockIdx.x;
    int w = blockIdx.y;
    int bh = blockIdx.z;

    int rowbase = half * 256 + wave * 32;  // q row within window [0,512)
    const u16* qb = qg + (((size_t)bh * 4096) + (size_t)w * 512 + rowbase) * 64;
    const u16* kb_ = kg + ((size_t)bh * 4096) * 64;
    const u16* vb = vT + ((size_t)bh * 64) * 4096;
    u16* pw = pls + wave * 32 * 40;

    // Resident q A-fragments: 2 m-tiles x 2 K-halves (DH=64)
    Frag qa[2][2];
    for (int mi = 0; mi < 2; ++mi)
        for (int kh = 0; kh < 2; ++kh) {
            const u16* p = qb + (size_t)(mi * 16 + l15) * 64 + kh * 32 + lh * 8;
            qa[mi][kh].q[0] = *(const uint4*)p;
            qa[mi][kh].q[1] = *(const uint4*)(p + 16);
        }

    f32x8 acc[2][4] = {};
    float mrow[2][8], lrow[2][8];
    for (int mi = 0; mi < 2; ++mi)
        for (int i = 0; i < 8; ++i) { mrow[mi][i] = -1e30f; lrow[mi][i] = 0.f; }

    int c0 = (w == 0) ? 16 : 0;
    int qr_max = rowbase + 31;
    int c_end = ((qr_max + 512) >> 5) + 1;
    if (c_end > 32) c_end = 32;

    for (int c = c0; c < c_end; ++c) {
        int t0 = w * 512 - 512 + c * 32;  // >= 0 on every executed chunk

        // s = q @ k^T for 32 keys (2 n-tiles), K = DH = 64 (2 wmma halves)
        f32x8 s[2][2] = {};
        for (int kh = 0; kh < 2; ++kh) {
            for (int ni = 0; ni < 2; ++ni) {
                Frag bk;
                const u16* p = kb_ + (size_t)(t0 + ni * 16 + l15) * 64 + kh * 32 + lh * 8;
                bk.q[0] = *(const uint4*)p;
                bk.q[1] = *(const uint4*)(p + 16);
                for (int mi = 0; mi < 2; ++mi)
                    s[mi][ni] = wmma_bf16(qa[mi][kh].v, bk.v, s[mi][ni]);
            }
        }

        // per-row chunk max; masking only on boundary chunks (wave-uniform test)
        float cm[2][8];
        bool need_mask = !((w > 0 || c >= 16) && (c * 32 + 31 <= rowbase + 512));
        if (need_mask) {
            for (int mi = 0; mi < 2; ++mi) {
                int qr0 = rowbase + mi * 16 + lh * 8;
                for (int i = 0; i < 8; ++i) {
                    int qr = qr0 + i;
                    float best = -1e30f;
                    for (int ni = 0; ni < 2; ++ni) {
                        int j = c * 32 + ni * 16 + l15;
                        float sv = s[mi][ni][i];
                        bool ok = (w > 0 || j >= 512) && (j <= qr + 512);
                        sv = ok ? sv : -1e30f;
                        s[mi][ni][i] = sv;
                        best = fmaxf(best, sv);
                    }
                    cm[mi][i] = best;
                }
            }
        } else {
            for (int mi = 0; mi < 2; ++mi)
                for (int i = 0; i < 8; ++i)
                    cm[mi][i] = fmaxf(s[mi][0][i], s[mi][1][i]);
        }
        for (int off = 1; off <= 8; off <<= 1)
            for (int mi = 0; mi < 2; ++mi)
                for (int i = 0; i < 8; ++i)
                    cm[mi][i] = fmaxf(cm[mi][i], __shfl_xor(cm[mi][i], off, 32));

        // online softmax rescale
        float sc[2][8];
        for (int mi = 0; mi < 2; ++mi)
            for (int i = 0; i < 8; ++i) {
                float mn = fmaxf(mrow[mi][i], cm[mi][i]);
                sc[mi][i] = __expf(mrow[mi][i] - mn);
                mrow[mi][i] = mn;
                lrow[mi][i] *= sc[mi][i];
            }
        for (int mi = 0; mi < 2; ++mi)
            for (int di = 0; di < 4; ++di)
                for (int i = 0; i < 8; ++i) acc[mi][di][i] *= sc[mi][i];

        // p = exp(s - m), stage to LDS (D-layout -> A-layout reshape)
        float ps[2][8] = {};
        for (int mi = 0; mi < 2; ++mi)
            for (int ni = 0; ni < 2; ++ni)
                for (int i = 0; i < 8; ++i) {
                    float p = __expf(s[mi][ni][i] - mrow[mi][i]);
                    ps[mi][i] += p;
                    pw[(mi * 16 + lh * 8 + i) * 40 + ni * 16 + l15] = f2b(p);
                }
        for (int off = 1; off <= 8; off <<= 1)
            for (int mi = 0; mi < 2; ++mi)
                for (int i = 0; i < 8; ++i) ps[mi][i] += __shfl_xor(ps[mi][i], off, 32);
        for (int mi = 0; mi < 2; ++mi)
            for (int i = 0; i < 8; ++i) lrow[mi][i] += ps[mi][i];

        Frag pa[2];
        for (int mi = 0; mi < 2; ++mi) {
            const u16* p = pw + (mi * 16 + l15) * 40 + lh * 8;
            pa[mi].q[0] = *(const uint4*)p;
            pa[mi].q[1] = *(const uint4*)(p + 16);
        }

        // o += p @ v  (v pre-transposed [d][token] => B column-major directly)
        for (int di = 0; di < 4; ++di) {
            Frag vf;
            const u16* p = vb + (size_t)(di * 16 + l15) * 4096 + t0 + lh * 8;
            vf.q[0] = *(const uint4*)p;
            vf.q[1] = *(const uint4*)(p + 16);
            for (int mi = 0; mi < 2; ++mi)
                acc[mi][di] = wmma_bf16(pa[mi].v, vf.v, acc[mi][di]);
        }
    }

    // normalize + store
    int b = bh >> 4, h = bh & 15;
    size_t tokbase = (size_t)b * 4096 + (size_t)w * 512 + rowbase;
    for (int mi = 0; mi < 2; ++mi)
        for (int di = 0; di < 4; ++di)
            for (int i = 0; i < 8; ++i) {
                size_t tok = tokbase + mi * 16 + lh * 8 + i;
                float ov = acc[mi][di][i] / lrow[mi][i];
                og[tok * 1024 + h * 64 + di * 16 + l15] = f2b(ov);
            }
}

// ---------------------------------------------------------------------------
// GEGLU: u bf16 [8192, 5504] (a | pad | g | pad) -> ga bf16 [8192, 2752]
// ---------------------------------------------------------------------------
__global__ __launch_bounds__(256) void geglu_kernel(const u16* __restrict__ u,
                                                    u16* __restrict__ ga) {
    size_t idx = (size_t)blockIdx.x * 256 + threadIdx.x;  // < 8192*2752
    size_t m = idx / 2752;
    int j = (int)(idx % 2752);
    float a = b2f(u[m * 5504 + j]);
    float g = b2f(u[m * 5504 + 2752 + j]);
    float gel = 0.5f * g * (1.0f + erff(g * 0.70710678118654752f));
    float r = (j < 2730) ? a * gel : 0.0f;
    ga[m * 2752 + j] = f2b(r);
}

// ---------------------------------------------------------------------------
extern "C" void kernel_launch(void* const* d_in, const int* in_sizes, int n_in,
                              void* d_out, int out_size, void* d_ws, size_t ws_size,
                              hipStream_t stream) {
    const float* x = (const float*)d_in[0];
    const float* ln1w = (const float*)d_in[1];
    const float* ln1b = (const float*)d_in[2];
    const float* wqkv = (const float*)d_in[3];
    const float* wout = (const float*)d_in[4];
    const float* ln2w = (const float*)d_in[5];
    const float* wff1 = (const float*)d_in[6];
    const float* wff2 = (const float*)d_in[7];
    float* out = (float*)d_out;

    char* ws = (char*)d_ws;
    size_t off = 0;
    auto alloc = [&](size_t bytes) -> void* {
        void* p = ws + off;
        off += (bytes + 255) & ~(size_t)255;
        return p;
    };

    const size_t BN = 8192;  // B * N
    u16* ubuf = (u16*)alloc(BN * 5504 * 2);   // FF1 output; also aliases qkv_bf16
    u16* qkvb = ubuf;                         // [8192, 3072] bf16 (dead before ubuf written)
    u16* h1 = (u16*)alloc(BN * 2752 * 2);     // LN1 out bf16; region reused as ga
    u16* ga = h1;
    u16* wqkvT = (u16*)alloc((size_t)3072 * 1024 * 2);
    u16* qg = (u16*)alloc((size_t)32 * 4096 * 64 * 2);
    u16* kg = (u16*)alloc((size_t)32 * 4096 * 64 * 2);
    u16* vT = (u16*)alloc((size_t)32 * 4096 * 64 * 2);
    u16* obuf = (u16*)alloc(BN * 1024 * 2);
    u16* woutT = (u16*)alloc((size_t)1024 * 1024 * 2);
    u16* h2 = (u16*)alloc(BN * 1024 * 2);
    u16* wff1T = (u16*)alloc((size_t)5504 * 1024 * 2);
    u16* wff2T = (u16*)alloc((size_t)1024 * 2752 * 2);

    // --- weight transposes / bf16 converts (padded) ---
    convT_kernel<<<(3072 * 1024) / 256, 256, 0, stream>>>(wqkv, 3072, 1024, 3072, wqkvT, 1024, 3072);
    convT_kernel<<<(1024 * 1024) / 256, 256, 0, stream>>>(wout, 1024, 1024, 1024, woutT, 1024, 1024);
    convT_kernel<<<(2752 * 1024) / 256, 256, 0, stream>>>(wff1, 5460, 1024, 2730, wff1T, 1024, 2752);
    convT_kernel<<<(2752 * 1024) / 256, 256, 0, stream>>>(wff1 + 2730, 5460, 1024, 2730,
                                                          wff1T + (size_t)2752 * 1024, 1024, 2752);
    convT_kernel<<<(1024 * 2752) / 256, 256, 0, stream>>>(wff2, 1024, 2730, 1024, wff2T, 2752, 1024);

    // --- attention branch ---
    ln_kernel<<<8192, 256, 0, stream>>>(x, ln1w, ln1b, 1, h1);
    gemm_bf16_kernel<1><<<dim3(24, 64), 256, 0, stream>>>(h1, 1024, wqkvT, 1024,
                                                          nullptr, qkvb, 3072, nullptr,
                                                          8192, 3072, 1024);
    rotary_kernel<<<32768, 256, 0, stream>>>(qkvb, qg, kg, vT);
    attn_kernel<<<dim3(2, 8, 32), 256, 0, stream>>>(qg, kg, vT, obuf);
    gemm_bf16_kernel<2><<<dim3(8, 64), 256, 0, stream>>>(obuf, 1024, woutT, 1024,
                                                         out, nullptr, 1024, x,
                                                         8192, 1024, 1024);  // out = o@Wo + x

    // --- feed-forward branch ---
    ln_kernel<<<8192, 256, 0, stream>>>(out, ln2w, nullptr, 0, h2);
    gemm_bf16_kernel<1><<<dim3(43, 64), 256, 0, stream>>>(h2, 1024, wff1T, 1024,
                                                          nullptr, ubuf, 5504, nullptr,
                                                          8192, 5504, 1024);
    geglu_kernel<<<(8192 * 2752) / 256, 256, 0, stream>>>(ubuf, ga);
    gemm_bf16_kernel<2><<<dim3(8, 64), 256, 0, stream>>>(ga, 2752, wff2T, 2752,
                                                         out, nullptr, 1024, out,
                                                         8192, 1024, 2752);  // out += ga@W2
}
